// CrossAttention_54717883351624
// MI455X (gfx1250) — matrix operand order repaired and checked by reading, not compiled, not run on hardware
//
#include <hip/hip_runtime.h>
#include <hip/hip_bf16.h>

// ---------------------------------------------------------------------------
// CrossAttention for MI455X (gfx1250, wave32, WMMA 16x16x32 f16)
//
// Pipeline:
//   1) convert x1/x2 and Wq/Wk/Wv/Wo fp32 -> f16 in workspace
//   2) Q = x1 @ Wq^T, K = x2 @ Wk^T, V = x2 @ Wv^T   (async-LDS WMMA GEMM)
//   3) per-token attention: S = K^T_h Q / 8, softmax over d, O = V @ attn
//   4) Y = O @ Wo^T + bo (WMMA GEMM, fp32 out)
//
// GEMM staging uses gfx1250 GLOBAL_LOAD_ASYNC_TO_LDS_B128 (ASYNCcnt) so tiles
// stream global->LDS without a VGPR round-trip; one s_wait_asynccnt 0 before
// the work-group barrier publishes the tile.
// ---------------------------------------------------------------------------

typedef __attribute__((ext_vector_type(16))) _Float16 v16h;
typedef __attribute__((ext_vector_type(8)))  _Float16 v8h;
typedef __attribute__((ext_vector_type(4)))  _Float16 v4h;
typedef __attribute__((ext_vector_type(8)))  float    v8f;

static __device__ inline v8f wmma_f16(v16h a, v16h b, v8f c) {
  // D = A(16x32 f16) * B(32x16 f16) + C(16x16 f32)
  return __builtin_amdgcn_wmma_f32_16x16x32_f16(
      /*neg_a=*/false, a, /*neg_b=*/false, b,
      /*c_mod=*/(short)0, c, /*reuse_a=*/false, /*reuse_b=*/false);
}

static __device__ inline v16h cat16(v8h lo, v8h hi) {
  v16h r;
#pragma unroll
  for (int i = 0; i < 8; ++i) { r[i] = lo[i]; r[i + 8] = hi[i]; }
  return r;
}

// 32 bytes global -> LDS, async (two b128); IOFFSET applies to both addresses
static __device__ inline void async_copy32(const _Float16* gsrc, _Float16* lds) {
  const unsigned la = (unsigned)(size_t)lds;           // addr[31:0] == LDS addr
  const unsigned long long ga = (unsigned long long)(size_t)gsrc;
  asm volatile(
      "global_load_async_to_lds_b128 %0, %1, off\n\t"
      "global_load_async_to_lds_b128 %0, %1, off offset:16"
      :: "v"(la), "v"(ga) : "memory");
}
static __device__ inline void async_copy16(const _Float16* gsrc, _Float16* lds) {
  const unsigned la = (unsigned)(size_t)lds;
  const unsigned long long ga = (unsigned long long)(size_t)gsrc;
  asm volatile("global_load_async_to_lds_b128 %0, %1, off"
               :: "v"(la), "v"(ga) : "memory");
}
static __device__ inline void async_wait0() {
  asm volatile("s_wait_asynccnt 0" ::: "memory");
}

// ---------------------------------------------------------------------------
// fp32 -> f16 conversion (vectorized x4)
// ---------------------------------------------------------------------------
__global__ void f32_to_f16_kernel(const float* __restrict__ s,
                                  _Float16* __restrict__ d, int n4) {
  int i = blockIdx.x * 256 + threadIdx.x;
  if (i < n4) {
    float4 f = ((const float4*)s)[i];
    v4h v;
    v[0] = (_Float16)f.x; v[1] = (_Float16)f.y;
    v[2] = (_Float16)f.z; v[3] = (_Float16)f.w;
    ((v4h*)d)[i] = v;
  }
}

// ---------------------------------------------------------------------------
// GEMM: C[M,N] = A[M,K] * Bw[N,K]^T (+bias).  M%128==0, N%128==0, K%32==0.
// Block: 256 threads (8 waves). Block tile 128(M) x 128(N), BK=32.
// Wave (wm = wave&3, wn = wave>>2) computes rows [32*wm, 32*wm+32) x cols
// [64*wn, 64*wn+64): 2x4 grid of 16x16 accumulators (8 WMMAs per 12
// ds_load_b128 per k-step).
//
// ISA 7.12.2 fragment layouts (wave32, 16-bit):
//   A 16x32: lane L (g=L/16, m=L%16) half j -> k = (j%8) + 8*g + 16*(j/8)
//            => halves 0..7 = k[8g..8g+7], halves 8..15 = k[16+8g..16+8g+7]
//   B 32x16: lane L (g=L/16, n=L%16) half j -> k = (j%8) + 8*g + 16*(j/8)
//   C 16x16 f32: vgpr r, lane L -> M = r + 8*(L/16), N = L%16
// ---------------------------------------------------------------------------
template <bool OUT_IS_F16>
__global__ __launch_bounds__(256) void gemm_wmma_kernel(
    const _Float16* __restrict__ A, const _Float16* __restrict__ Bw,
    void* __restrict__ Cptr, const float* __restrict__ bias,
    int M, int N, int K) {
  const int tid  = threadIdx.x;
  const int wave = tid >> 5;
  const int lane = tid & 31;
  const int g    = lane >> 4;
  const int ln   = lane & 15;
  const int wm   = wave & 3;   // 0..3 -> 32-row M strip
  const int wn   = wave >> 2;  // 0..1 -> 64-col N half
  const int blockM = blockIdx.y * 128;
  const int blockN = blockIdx.x * 128;

  // +8 half padding keeps rows 16B-aligned (80B stride) and spreads banks
  __shared__ __align__(16) _Float16 Asub[128][40];
  __shared__ __align__(16) _Float16 Bsub[128][40];

  v8f acc[2][4];
#pragma unroll
  for (int i = 0; i < 2; ++i)
#pragma unroll
    for (int tn = 0; tn < 4; ++tn) acc[i][tn] = v8f{};

  // staging: each thread covers 16 k-elements of one row (A and B alike)
  const int sr = tid >> 1, sc = (tid & 1) * 16;
  const _Float16* arow = &A[(size_t)(blockM + sr) * K + sc];
  const _Float16* brow = &Bw[(size_t)(blockN + sr) * K + sc];

  for (int k0 = 0; k0 < K; k0 += 32) {
    // ---- stage A/B tiles: async global -> LDS (no VGPR round-trip) ----
    async_copy32(arow + k0, &Asub[sr][sc]);
    async_copy32(brow + k0, &Bsub[sr][sc]);
    if (k0 + 32 < K) {
      __builtin_prefetch(arow + k0 + 32, 0, 3);
      __builtin_prefetch(brow + k0 + 32, 0, 3);
    }
    async_wait0();
    __syncthreads();

    // ---- A fragments: 2 per wave (two ds_load_b128 each) ----
    v16h af[2];
#pragma unroll
    for (int i = 0; i < 2; ++i) {
      const int row = 32 * wm + 16 * i + ln;
      af[i] = cat16(*(const v8h*)&Asub[row][8 * g],
                    *(const v8h*)&Asub[row][16 + 8 * g]);
    }
    // ---- B fragments: 4 per wave; 8 WMMAs reuse af/bf ----
#pragma unroll
    for (int tn = 0; tn < 4; ++tn) {
      const _Float16* pb = &Bsub[64 * wn + 16 * tn + ln][8 * g];
      v16h bf = cat16(*(const v8h*)pb, *(const v8h*)(pb + 16));
      acc[0][tn] = wmma_f16(af[0], bf, acc[0][tn]);
      acc[1][tn] = wmma_f16(af[1], bf, acc[1][tn]);
    }
    __syncthreads();
  }

  // ---- write C per the 16x16 f32 layout ----
#pragma unroll
  for (int i = 0; i < 2; ++i) {
#pragma unroll
    for (int tn = 0; tn < 4; ++tn) {
      const int col = blockN + 64 * wn + 16 * tn + ln;
#pragma unroll
      for (int r = 0; r < 8; ++r) {
        const int row = blockM + 32 * wm + 16 * i + r + 8 * g;
        if constexpr (OUT_IS_F16) {
          ((_Float16*)Cptr)[(size_t)row * N + col] = (_Float16)acc[i][tn][r];
        } else {
          ((float*)Cptr)[(size_t)row * N + col] = acc[i][tn][r] + bias[col];
        }
      }
    }
  }
}

// ---------------------------------------------------------------------------
// Per-token attention.  One block (4 waves) per token t.
// Q/K/V global layout: [t][h*64 + c] f16 (h<16, c<64).
//
// LDS staging (transposed, head dim padded 16->32 with zeros so fragment
// loads are branch-free b128s):
//   Kt[c][32] = K[h][c]   (h in 0..15; 16..31 zero)   -> A fragments of S
//   Qt[d][32] = Q[h][d]   (padded)                    -> B fragments of S
//   Vs[h][64]  (async row-major copy)                  -> A fragments of O
//   Att[d][72] = attn[v][d] transposed                 -> B fragments of O
//
// Wave w owns S rows c in [16w,16w+16) over all 64 d; softmax over d lives
// entirely inside a 16-lane half-group (C layout: M = r + 8*(lane/16)).
// ---------------------------------------------------------------------------
__global__ __launch_bounds__(128) void attn_kernel(
    const _Float16* __restrict__ Q, const _Float16* __restrict__ Kc,
    const _Float16* __restrict__ V, _Float16* __restrict__ O) {
  const int t    = blockIdx.x;
  const int tid  = threadIdx.x;
  const int wave = tid >> 5;
  const int lane = tid & 31;
  const int g    = lane >> 4;
  const int ln   = lane & 15;

  __shared__ __align__(16) _Float16 Qt[64][32];   // [d][h], padded
  __shared__ __align__(16) _Float16 Kt[64][32];   // [c][h], padded
  __shared__ __align__(16) _Float16 Vs[16][64];   // [h][v]
  __shared__ __align__(16) _Float16 Att[64][72];  // [d][v]

  // ---- stage Q/K/V ----
  {
    const size_t base = (size_t)t * 1024;
    const int h  = tid >> 3;        // 0..15
    const int d0 = (tid & 7) * 8;   // 0..56
    // V: row-major copy -> async straight into LDS
    async_copy16(V + base + h * 64 + d0, &Vs[h][d0]);
    // Q/K need a transpose: through VGPRs
    v8h q = *(const v8h*)(Q + base + h * 64 + d0);
    v8h k = *(const v8h*)(Kc + base + h * 64 + d0);
#pragma unroll
    for (int i = 0; i < 8; ++i) {
      Qt[d0 + i][h] = q[i];
      Kt[d0 + i][h] = k[i];
    }
    // zero the padded upper half of the head dimension
    v8h z = {};
    *(v8h*)&Qt[tid >> 1][16 + (tid & 1) * 8] = z;
    *(v8h*)&Kt[tid >> 1][16 + (tid & 1) * 8] = z;
    async_wait0();
  }
  __syncthreads();

  // ---- S = K^T Q for 4 d-blocks (all fragment loads are b128, uniform) ----
  const int crow = 16 * wave + ln;  // this wave's S row (A fragment M index)
  v16h af = cat16(*(const v8h*)&Kt[crow][8 * g],
                  *(const v8h*)&Kt[crow][16 + 8 * g]);
  v8f sacc[4];
#pragma unroll
  for (int tb = 0; tb < 4; ++tb) {
    const int d = 16 * tb + ln;
    v16h bf = cat16(*(const v8h*)&Qt[d][8 * g],
                    *(const v8h*)&Qt[d][16 + 8 * g]);
    v8f z = {};
    sacc[tb] = wmma_f16(af, bf, z);
  }

  // ---- softmax over d (rows live in one 16-lane half-group) ----
  const float scale = 0.125f;  // 1/sqrt(64)
#pragma unroll
  for (int r = 0; r < 8; ++r) {
    float mx = -3.4e38f;
#pragma unroll
    for (int tb = 0; tb < 4; ++tb) mx = fmaxf(mx, sacc[tb][r] * scale);
#pragma unroll
    for (int m = 1; m < 16; m <<= 1) mx = fmaxf(mx, __shfl_xor(mx, m, 32));
    float sum = 0.f;
#pragma unroll
    for (int tb = 0; tb < 4; ++tb) {
      float e = __expf(sacc[tb][r] * scale - mx);
      sacc[tb][r] = e;
      sum += e;
    }
#pragma unroll
    for (int m = 1; m < 16; m <<= 1) sum += __shfl_xor(sum, m, 32);
    const float inv = 1.f / sum;
    const int c = 16 * wave + r + 8 * g;  // attn row v = c
#pragma unroll
    for (int tb = 0; tb < 4; ++tb)
      Att[16 * tb + ln][c] = (_Float16)(sacc[tb][r] * inv);  // transposed
  }
  __syncthreads();

  // ---- O[:, d-block=wave] = V(16x64) @ attn(64x16), two k=32 steps ----
  const int dcol = 16 * wave + ln;
  v8f oacc = {};
#pragma unroll
  for (int s = 0; s < 2; ++s) {
    // A[h, v]: contiguous runs of Vs row h=ln (two ds_load_b128)
    v16h va = cat16(*(const v8h*)&Vs[ln][32 * s + 8 * g],
                    *(const v8h*)&Vs[ln][32 * s + 16 + 8 * g]);
    // B[v, d]: half j -> v = 32s + 8g + (j%8) + 16*(j/8), matching A
    v16h vb = cat16(*(const v8h*)&Att[dcol][32 * s + 8 * g],
                    *(const v8h*)&Att[dcol][32 * s + 16 + 8 * g]);
    oacc = wmma_f16(va, vb, oacc);
  }

  // ---- store O[t][h*64 + d] ----
  {
    const size_t base = (size_t)t * 1024;
#pragma unroll
    for (int r = 0; r < 8; ++r) {
      const int h = r + 8 * g;
      O[base + h * 64 + dcol] = (_Float16)oacc[r];
    }
  }
}

// ---------------------------------------------------------------------------
// Launch
// ---------------------------------------------------------------------------
extern "C" void kernel_launch(void* const* d_in, const int* in_sizes, int n_in,
                              void* d_out, int out_size, void* d_ws, size_t ws_size,
                              hipStream_t stream) {
  (void)in_sizes; (void)n_in; (void)out_size; (void)ws_size;
  const float* x1 = (const float*)d_in[0];
  const float* x2 = (const float*)d_in[1];
  const float* Wq = (const float*)d_in[2];
  const float* Wk = (const float*)d_in[3];
  const float* Wv = (const float*)d_in[4];
  const float* Wo = (const float*)d_in[5];
  const float* bo = (const float*)d_in[6];
  float* out = (float*)d_out;

  const int Mtok = 8 * 2048;  // 16384 tokens
  const int D    = 1024;

  // workspace (f16): 4 weights (2 MiB) + x1/x2 (32 MiB) + Q/K/V/O (32 MiB)
  char* ws = (char*)d_ws;
  _Float16* Wq16 = (_Float16*)ws; ws += (size_t)D * D * 2;
  _Float16* Wk16 = (_Float16*)ws; ws += (size_t)D * D * 2;
  _Float16* Wv16 = (_Float16*)ws; ws += (size_t)D * D * 2;
  _Float16* Wo16 = (_Float16*)ws; ws += (size_t)D * D * 2;
  _Float16* X1h  = (_Float16*)ws; ws += (size_t)Mtok * D * 2;
  _Float16* X2h  = (_Float16*)ws; ws += (size_t)Mtok * D * 2;
  _Float16* Q16  = (_Float16*)ws; ws += (size_t)Mtok * D * 2;
  _Float16* K16  = (_Float16*)ws; ws += (size_t)Mtok * D * 2;
  _Float16* V16  = (_Float16*)ws; ws += (size_t)Mtok * D * 2;
  _Float16* O16  = (_Float16*)ws; ws += (size_t)Mtok * D * 2;

  // 1) fp32 -> f16 conversions (weights + activations)
  {
    const int wn4 = D * D / 4;
    const int wgrid = (wn4 + 255) / 256;
    f32_to_f16_kernel<<<wgrid, 256, 0, stream>>>(Wq, Wq16, wn4);
    f32_to_f16_kernel<<<wgrid, 256, 0, stream>>>(Wk, Wk16, wn4);
    f32_to_f16_kernel<<<wgrid, 256, 0, stream>>>(Wv, Wv16, wn4);
    f32_to_f16_kernel<<<wgrid, 256, 0, stream>>>(Wo, Wo16, wn4);
    const int xn4 = Mtok * D / 4;
    const int xgrid = (xn4 + 255) / 256;
    f32_to_f16_kernel<<<xgrid, 256, 0, stream>>>(x1, X1h, xn4);
    f32_to_f16_kernel<<<xgrid, 256, 0, stream>>>(x2, X2h, xn4);
  }

  // 2) projections
  dim3 ggrid(D / 128, Mtok / 128);  // (8, 128)
  gemm_wmma_kernel<true><<<ggrid, 256, 0, stream>>>(X1h, Wq16, Q16, nullptr, Mtok, D, D);
  gemm_wmma_kernel<true><<<ggrid, 256, 0, stream>>>(X2h, Wk16, K16, nullptr, Mtok, D, D);
  gemm_wmma_kernel<true><<<ggrid, 256, 0, stream>>>(X2h, Wv16, V16, nullptr, Mtok, D, D);

  // 3) per-token attention
  attn_kernel<<<Mtok, 128, 0, stream>>>(Q16, K16, V16, O16);

  // 4) output projection + bias (fp32 out)
  gemm_wmma_kernel<false><<<ggrid, 256, 0, stream>>>(O16, Wo16, out, bo, Mtok, D, D);
}
